// AttentionBlock_76690936037520
// MI455X (gfx1250) — matrix operand order repaired and checked by reading, not compiled, not run on hardware
//
#include <hip/hip_runtime.h>
#include <hip/hip_bf16.h>
#include <math.h>

// ---------------------------------------------------------------------------
// AttentionBlock for MI455X (gfx1250, wave32, WMMA).
// B=4, H=W=64 (N=4096 tokens), C=512, G=32 groups (16 ch/group).
// GN -> bf16 tokens -> WMMA GEMM Q/K/V (V stored transposed) ->
// flash-attention (WMMA, online softmax, O in LDS) -> WMMA GEMM + bias + resid.
// All WMMA operand loads are contiguous b128 pairs (weights pre-transposed,
// V stored [B,C,N]); no strided scalar-load storms.
// Workspace usage ~86 MB.
// ---------------------------------------------------------------------------

typedef __attribute__((ext_vector_type(16))) __bf16 v16bf;
typedef __attribute__((ext_vector_type(8)))  float  v8f;

#define BDIM 4
#define NTOK 4096
#define CDIM 512
#define MROWS (BDIM * NTOK)   // 16384

__device__ inline __bf16 f2bf(float f) { return (__bf16)f; }

__device__ inline v8f wmma_bf16(v16bf a, v16bf b, v8f c) {
  // D = A(16x32 bf16) * B(32x16 bf16) + C(16x16 f32)
  return __builtin_amdgcn_wmma_f32_16x16x32_bf16(
      /*neg_a=*/false, a, /*neg_b=*/false, b,
      /*c_mod=*/(short)0, c, /*reuse_a=*/false, /*reuse_b=*/false);
}

// A-fragment, 16x32 bf16 tile, row-major source with leading dim `ld`.
// Lane layout (ISA 7.12.2): row m = lane&15, group g = lane>>4,
// halves 0..7  -> k = g*8 + 0..7 ; halves 8..15 -> k = 16 + g*8 + 0..7.
__device__ inline v16bf load_a_frag(const __bf16* p, int ld) {
  int lane = threadIdx.x & 31;
  int m = lane & 15, g = lane >> 4;
  const __bf16* row = p + (size_t)m * ld;
  v16bf a;
  *(uint4*)&a       = *(const uint4*)(row + g * 8);        // k = g*8 .. g*8+7
  *((uint4*)&a + 1) = *(const uint4*)(row + 16 + g * 8);   // k = 16+g*8 ..
  return a;
}

// B-fragment, 32x16 bf16 tile where B[k][n] = p[n*ld + k] (source stores the
// n-dimension as rows: K^T, W^T, V^T). Lane n = lane&15,
// halves 0..7 -> k = g*8 + 0..7 ; halves 8..15 -> k = 16 + g*8 + 0..7.
__device__ inline v16bf load_bT_frag(const __bf16* p, int ld) {
  int lane = threadIdx.x & 31;
  int n = lane & 15, g = lane >> 4;
  const __bf16* col = p + (size_t)n * ld;
  v16bf b;
  *(uint4*)&b       = *(const uint4*)(col + g * 8);
  *((uint4*)&b + 1) = *(const uint4*)(col + 16 + g * 8);
  return b;
}

// ---------------------------------------------------------------------------
// GroupNorm statistics: one block per (b, group). 65536 values reduced.
// ---------------------------------------------------------------------------
__global__ __launch_bounds__(256) void gn_stats_kernel(const float* __restrict__ x,
                                                       float* __restrict__ stats) {
  int b = blockIdx.x >> 5, g = blockIdx.x & 31;
  const float* xb = x + (size_t)b * NTOK * CDIM + g * 16;
  float s = 0.f, ss = 0.f;
  for (int i = threadIdx.x; i < NTOK * 16; i += 256) {
    int hw = i >> 4, c = i & 15;
    float v = xb[(size_t)hw * CDIM + c];
    s += v; ss += v * v;
  }
  __shared__ float rs[256], rss[256];
  rs[threadIdx.x] = s; rss[threadIdx.x] = ss;
  __syncthreads();
  for (int o = 128; o > 0; o >>= 1) {
    if (threadIdx.x < (unsigned)o) {
      rs[threadIdx.x]  += rs[threadIdx.x + o];
      rss[threadIdx.x] += rss[threadIdx.x + o];
    }
    __syncthreads();
  }
  if (threadIdx.x == 0) {
    float mean = rs[0] * (1.f / 65536.f);
    float var  = rss[0] * (1.f / 65536.f) - mean * mean;
    stats[blockIdx.x * 2 + 0] = mean;
    stats[blockIdx.x * 2 + 1] = rsqrtf(var + 1e-6f);
  }
}

// Normalize + affine, emit bf16 tokens t[B*N, C].
__global__ __launch_bounds__(256) void gn_apply_kernel(const float* __restrict__ x,
                                                       const float* __restrict__ stats,
                                                       const float* __restrict__ gamma,
                                                       const float* __restrict__ beta,
                                                       __bf16* __restrict__ t) {
  size_t idx = (size_t)blockIdx.x * 256 + threadIdx.x;
  if (idx >= (size_t)MROWS * CDIM) return;
  int c = (int)(idx & (CDIM - 1));
  int b = (int)(idx >> 21);            // / (4096*512)
  int g = c >> 4;
  float mean = stats[(b * 32 + g) * 2 + 0];
  float rstd = stats[(b * 32 + g) * 2 + 1];
  float v = (x[idx] - mean) * rstd * gamma[c] + beta[c];
  t[idx] = f2bf(v);
}

// Convert W[K,N] f32 -> W^T[N,K] bf16 (512x512 per matrix; one-time cost).
__global__ __launch_bounds__(256) void wt_bf16_kernel(const float* __restrict__ W,
                                                      __bf16* __restrict__ Wt) {
  int i = blockIdx.x * 256 + threadIdx.x;   // i = n*512 + k (dst index)
  if (i >= CDIM * CDIM) return;
  int n = i >> 9, k = i & (CDIM - 1);
  Wt[i] = f2bf(W[(size_t)k * CDIM + n]);
}

// ---------------------------------------------------------------------------
// WMMA GEMM: C[M,N] = A[M,K]bf16 * B[K,N]bf16 + bias, with B supplied as
// B^T[N,K] (row-major). Each wave computes a 16x64 strip (4 N-tiles),
// reusing one A fragment across 4 WMMAs per K-step. 8 waves / block.
// mode 0: bf16 [M,N] out.  mode 1: bf16 out transposed per batch -> [B][N][M']
// (i.e. Vt[b][col][tok]).  mode 2: f32 out + residual.
// ---------------------------------------------------------------------------
__global__ __launch_bounds__(256) void gemm_bias_kernel(const __bf16* __restrict__ A,
                                                        const __bf16* __restrict__ Bt,
                                                        const float* __restrict__ bias,
                                                        __bf16* __restrict__ Obf,
                                                        float* __restrict__ Of32,
                                                        const float* __restrict__ resid,
                                                        int M, int N, int K, int mode) {
  int wave = threadIdx.x >> 5;
  int tile = blockIdx.x * 8 + wave;
  int ntn = N >> 6;                       // 64-wide strips
  int mt = tile / ntn, nq = tile - mt * ntn;
  if (mt * 16 >= M) return;
  const __bf16* Ab = A + (size_t)mt * 16 * K;
  const __bf16* Bb = Bt + (size_t)nq * 64 * K;
  v8f acc0 = {}, acc1 = {}, acc2 = {}, acc3 = {};
  for (int k = 0; k < K; k += 32) {
    __builtin_prefetch(Ab + k + 128, 0, 1);  // global_prefetch_b8
    v16bf a  = load_a_frag(Ab + k, K);
    v16bf b0 = load_bT_frag(Bb + (size_t)0  * 16 * K + k, K);
    v16bf b1 = load_bT_frag(Bb + (size_t)1  * 16 * K + k, K);
    v16bf b2 = load_bT_frag(Bb + (size_t)2  * 16 * K + k, K);
    v16bf b3 = load_bT_frag(Bb + (size_t)3  * 16 * K + k, K);
    acc0 = wmma_bf16(a, b0, acc0);
    acc1 = wmma_bf16(a, b1, acc1);
    acc2 = wmma_bf16(a, b2, acc2);
    acc3 = wmma_bf16(a, b3, acc3);
  }
  int lane = threadIdx.x & 31, n = lane & 15, g = lane >> 4;
  v8f accs[4] = {acc0, acc1, acc2, acc3};
#pragma unroll
  for (int j = 0; j < 4; ++j) {
    int col = nq * 64 + j * 16 + n;
    float bv = bias[col];
#pragma unroll
    for (int v = 0; v < 8; ++v) {
      size_t row = (size_t)mt * 16 + v + 8 * g;
      float val = accs[j][v] + bv;
      if (mode == 2) {
        Of32[row * N + col] = val + resid[row * N + col];
      } else if (mode == 1) {
        size_t b = row >> 12, tok = row & (NTOK - 1);
        Obf[(b * CDIM + col) * NTOK + tok] = f2bf(val);   // Vt[b][col][tok]
      } else {
        Obf[row * N + col] = f2bf(val);
      }
    }
  }
}

// ---------------------------------------------------------------------------
// Flash attention: softmax(Q K^T / sqrt(C)) V, streamed over keys.
// Block = 4 waves (128 threads) handles 64 query rows; each wave owns 16 rows.
// Q fragments cached in VGPRs (16 x v16bf); O accumulator (64x512 f32) in
// dynamic LDS; P transposed D-layout -> A-layout through a per-wave LDS tile.
// V supplied transposed [B][C][N] so PV B-fragments are contiguous reads.
// ---------------------------------------------------------------------------
__global__ __launch_bounds__(128) void flash_attn_kernel(const __bf16* __restrict__ Q,
                                                         const __bf16* __restrict__ Kmat,
                                                         const __bf16* __restrict__ Vt,
                                                         __bf16* __restrict__ O) {
  extern __shared__ float sm[];
  float* ldsO = sm;                        // 64*512 f32 = 128 KB
  float* ldsP = sm + 64 * CDIM;            // 4 waves * 16*32 f32 = 8 KB
  float* ldsL = sm + 64 * CDIM + 4 * 512;  // 64 row sums

  int b  = blockIdx.x >> 6;          // 64 query-blocks per batch
  int qb = blockIdx.x & 63;
  int wave = threadIdx.x >> 5;
  int lane = threadIdx.x & 31;
  int nn = lane & 15, g = lane >> 4;

  const __bf16* Qb  = Q + ((size_t)b * NTOK + qb * 64 + wave * 16) * CDIM;
  const __bf16* Kb  = Kmat + (size_t)b * NTOK * CDIM;
  const __bf16* Vtb = Vt + (size_t)b * CDIM * NTOK;

  // Cache the wave's 16 query rows as A-fragments (16 K-steps of 32).
  v16bf aq[16];
#pragma unroll
  for (int kk = 0; kk < 16; ++kk) aq[kk] = load_a_frag(Qb + kk * 32, CDIM);

  float* myO = ldsO + wave * 16 * CDIM;
  for (int i = lane; i < 16 * CDIM; i += 32) myO[i] = 0.f;

  float mrow[8], lrow[8], srow[8];
#pragma unroll
  for (int v = 0; v < 8; ++v) { mrow[v] = -1e30f; lrow[v] = 0.f; }

  const float sscale = 0.044194173824159216f;  // 512^-0.5
  float* myP = ldsP + wave * 16 * 32;

  for (int kt = 0; kt < NTOK; kt += 32) {
    // S tile 16 rows x 32 keys as two 16x16 D fragments.
    v8f s0 = {}, s1 = {};
#pragma unroll
    for (int kk = 0; kk < 16; ++kk) {
      v16bf b0 = load_bT_frag(Kb + (size_t)(kt +  0) * CDIM + kk * 32, CDIM);
      v16bf b1 = load_bT_frag(Kb + (size_t)(kt + 16) * CDIM + kk * 32, CDIM);
      s0 = wmma_bf16(aq[kk], b0, s0);
      s1 = wmma_bf16(aq[kk], b1, s1);
    }
    // Online softmax update per row (row m = v + 8g, 16 lanes/row).
#pragma unroll
    for (int v = 0; v < 8; ++v) {
      float a0 = s0[v] * sscale, a1 = s1[v] * sscale;
      float rm = fmaxf(a0, a1);
      rm = fmaxf(rm, __shfl_xor(rm, 1));
      rm = fmaxf(rm, __shfl_xor(rm, 2));
      rm = fmaxf(rm, __shfl_xor(rm, 4));
      rm = fmaxf(rm, __shfl_xor(rm, 8));
      float mn = fmaxf(mrow[v], rm);
      srow[v] = __expf(mrow[v] - mn);
      mrow[v] = mn;
      float e0 = __expf(a0 - mn), e1 = __expf(a1 - mn);
      float rs = e0 + e1;
      rs += __shfl_xor(rs, 1);
      rs += __shfl_xor(rs, 2);
      rs += __shfl_xor(rs, 4);
      rs += __shfl_xor(rs, 8);
      lrow[v] = lrow[v] * srow[v] + rs;
      int r = v + 8 * g;
      myP[r * 32 + nn]      = e0;   // keys kt..kt+15
      myP[r * 32 + 16 + nn] = e1;   // keys kt+16..kt+31
    }
    // Read P back in A-fragment layout (wave-private LDS; DS ops in-order).
    v16bf ap;
#pragma unroll
    for (int t = 0; t < 8; ++t) ap[t]     = f2bf(myP[nn * 32 + g * 8 + t]);
#pragma unroll
    for (int t = 0; t < 8; ++t) ap[8 + t] = f2bf(myP[nn * 32 + 16 + g * 8 + t]);

    // O[16 x 512] += P(16x32) * V(32x512), rescaled by exp(m_old - m_new).
    // B-frag: B[k=key][n=ch] = Vt[ch][kt+key] -> contiguous per lane.
    for (int nb = 0; nb < 32; ++nb) {
      v16bf bv = load_bT_frag(Vtb + (size_t)nb * 16 * NTOK + kt, NTOK);
      v8f c;
#pragma unroll
      for (int v = 0; v < 8; ++v)
        c[v] = myO[(v + 8 * g) * CDIM + nb * 16 + nn] * srow[v];
      c = wmma_bf16(ap, bv, c);
#pragma unroll
      for (int v = 0; v < 8; ++v)
        myO[(v + 8 * g) * CDIM + nb * 16 + nn] = c[v];
    }
  }

  if (nn == 0) {
#pragma unroll
    for (int v = 0; v < 8; ++v) ldsL[wave * 16 + v + 8 * g] = lrow[v];
  }
  __syncthreads();

  __bf16* Ob = O + ((size_t)b * NTOK + qb * 64) * CDIM;
  for (int i = threadIdx.x; i < 64 * CDIM; i += 128) {
    int r = i >> 9;  // / CDIM
    Ob[i] = f2bf(ldsO[i] / ldsL[r]);
  }
}

// ---------------------------------------------------------------------------
extern "C" void kernel_launch(void* const* d_in, const int* in_sizes, int n_in,
                              void* d_out, int out_size, void* d_ws, size_t ws_size,
                              hipStream_t stream) {
  const float* x     = (const float*)d_in[0];
  const float* gamma = (const float*)d_in[1];
  const float* beta  = (const float*)d_in[2];
  const float* Wq = (const float*)d_in[3];
  const float* bq = (const float*)d_in[4];
  const float* Wk = (const float*)d_in[5];
  const float* bk = (const float*)d_in[6];
  const float* Wv = (const float*)d_in[7];
  const float* bv = (const float*)d_in[8];
  const float* Wo = (const float*)d_in[9];
  const float* bo = (const float*)d_in[10];
  float* out = (float*)d_out;

  // Workspace carve-up (~86 MB).
  char* ws = (char*)d_ws;
  size_t off = 0;
  float*  stats = (float*)(ws + off); off += 4096;
  __bf16* tb    = (__bf16*)(ws + off); off += (size_t)MROWS * CDIM * 2;
  __bf16* wqt   = (__bf16*)(ws + off); off += (size_t)CDIM * CDIM * 2;
  __bf16* wkt   = (__bf16*)(ws + off); off += (size_t)CDIM * CDIM * 2;
  __bf16* wvt   = (__bf16*)(ws + off); off += (size_t)CDIM * CDIM * 2;
  __bf16* wot   = (__bf16*)(ws + off); off += (size_t)CDIM * CDIM * 2;
  __bf16* Qb    = (__bf16*)(ws + off); off += (size_t)MROWS * CDIM * 2;
  __bf16* Kb    = (__bf16*)(ws + off); off += (size_t)MROWS * CDIM * 2;
  __bf16* Vtb   = (__bf16*)(ws + off); off += (size_t)MROWS * CDIM * 2;  // [B][C][N]
  __bf16* Ob    = (__bf16*)(ws + off); off += (size_t)MROWS * CDIM * 2;
  (void)ws_size; (void)in_sizes; (void)n_in; (void)out_size;

  // 1) GroupNorm stats + apply (bf16 tokens).
  gn_stats_kernel<<<BDIM * 32, 256, 0, stream>>>(x, stats);
  int nelem = MROWS * CDIM;
  gn_apply_kernel<<<(nelem + 255) / 256, 256, 0, stream>>>(x, stats, gamma, beta, tb);

  // 2) Weights -> transposed bf16.
  int wn = CDIM * CDIM;
  wt_bf16_kernel<<<(wn + 255) / 256, 256, 0, stream>>>(Wq, wqt);
  wt_bf16_kernel<<<(wn + 255) / 256, 256, 0, stream>>>(Wk, wkt);
  wt_bf16_kernel<<<(wn + 255) / 256, 256, 0, stream>>>(Wv, wvt);
  wt_bf16_kernel<<<(wn + 255) / 256, 256, 0, stream>>>(Wo, wot);

  // 3) Q/K/V projections (16384x512x512, WMMA). 16x64 strip per wave.
  int nstrip = (MROWS / 16) * (CDIM / 64);   // 8192 waves
  int gblocks = (nstrip + 7) / 8;            // 1024 blocks
  gemm_bias_kernel<<<gblocks, 256, 0, stream>>>(tb, wqt, bq, Qb, nullptr, nullptr,
                                                MROWS, CDIM, CDIM, 0);
  gemm_bias_kernel<<<gblocks, 256, 0, stream>>>(tb, wkt, bk, Kb, nullptr, nullptr,
                                                MROWS, CDIM, CDIM, 0);
  gemm_bias_kernel<<<gblocks, 256, 0, stream>>>(tb, wvt, bv, Vtb, nullptr, nullptr,
                                                MROWS, CDIM, CDIM, 1);

  // 4) Flash attention. Dynamic LDS: 64x512 f32 O + 4x16x32 P + 64 row sums.
  size_t smem = (size_t)(64 * CDIM + 4 * 512 + 64) * sizeof(float);  // ~136 KB
  flash_attn_kernel<<<BDIM * (NTOK / 64), 128, smem, stream>>>(Qb, Kb, Vtb, Ob);

  // 5) Output projection + bias + residual, f32 to d_out.
  gemm_bias_kernel<<<gblocks, 256, 0, stream>>>(Ob, wot, bo, nullptr, out, x,
                                                MROWS, CDIM, CDIM, 2);
}
